// MambaBlock_8684423873294
// MI455X (gfx1250) — matrix operand order repaired and checked by reading, not compiled, not run on hardware
//
#include <hip/hip_runtime.h>

// ---------------- problem constants ----------------
constexpr int D_MODEL = 1024;
constexpr int D_INNER = 2048;
constexpr int D_STATE = 16;
constexpr int D_CONV  = 4;
constexpr int DT_RANK = 64;
constexpr int BATCH   = 4;
constexpr int SEQLEN  = 2048;
constexpr int MROWS   = BATCH * SEQLEN;          // 8192
constexpr int XPROJ_N = DT_RANK + 2 * D_STATE;   // 96
constexpr int LCHUNK  = 128;                     // scan LDS staging chunk

// ---------------- types ----------------
typedef __attribute__((ext_vector_type(16))) __bf16   bf16x16;
typedef __attribute__((ext_vector_type(8)))  float    v8f;
typedef __attribute__((ext_vector_type(4)))  unsigned u32x4;

union Frag { bf16x16 v; u32x4 q[2]; };

__device__ __forceinline__ __bf16 f2bf(float f) {
    unsigned u = __builtin_bit_cast(unsigned, f);
    unsigned r = u + 0x7FFFu + ((u >> 16) & 1u);       // round-to-nearest-even
    unsigned short h = (unsigned short)(r >> 16);
    return __builtin_bit_cast(__bf16, h);
}
__device__ __forceinline__ float bf2f(__bf16 b) {
    unsigned short h = __builtin_bit_cast(unsigned short, b);
    unsigned u = ((unsigned)h) << 16;
    return __builtin_bit_cast(float, u);
}
__device__ __forceinline__ float sigmoidf_(float x) { return 1.f / (1.f + __expf(-x)); }

// ---------------- elementwise converts ----------------
__global__ void f32_to_bf16_kernel(const float* __restrict__ s, __bf16* __restrict__ d, int n) {
    for (int i = blockIdx.x * blockDim.x + threadIdx.x; i < n; i += gridDim.x * blockDim.x)
        d[i] = f2bf(s[i]);
}

// x_dbl[:, 0:DT_RANK] (row stride 96) -> bf16 [MROWS, DT_RANK]
__global__ void slice_dt_bf16_kernel(const float* __restrict__ src, __bf16* __restrict__ dst) {
    int n = MROWS * DT_RANK;
    for (int i = blockIdx.x * blockDim.x + threadIdx.x; i < n; i += gridDim.x * blockDim.x) {
        int r = i >> 6, c = i & 63;
        dst[i] = f2bf(src[(size_t)r * XPROJ_N + c]);
    }
}

// ---------------- RMSNorm -> bf16 ----------------
__global__ __launch_bounds__(256)
void rmsnorm_kernel(const float* __restrict__ x, const float* __restrict__ w,
                    __bf16* __restrict__ xn) {
    __shared__ float red[256];
    const int row = blockIdx.x;
    const float* xr = x + (size_t)row * D_MODEL;
    float ss = 0.f;
    for (int c = threadIdx.x; c < D_MODEL; c += 256) { float v = xr[c]; ss += v * v; }
    red[threadIdx.x] = ss;
    __syncthreads();
    for (int s = 128; s > 0; s >>= 1) {
        if ((int)threadIdx.x < s) red[threadIdx.x] += red[threadIdx.x + s];
        __syncthreads();
    }
    const float scale = rsqrtf(red[0] * (1.f / D_MODEL) + 1.1920929e-07f);
    for (int c = threadIdx.x; c < D_MODEL; c += 256)
        xn[(size_t)row * D_MODEL + c] = f2bf(xr[c] * scale * w[c]);
}

// ---------------- WMMA bf16 GEMM: C[M,N] = A[M,K] * W[N,K]^T ----------------
// Wave tile = (16*RM) x (16*RN): RM*RN WMMAs per 32-wide K-step, RM A-frags,
// RN B-frags (reuse factors RN and RM). Fragments per documented CDNA5 layouts.
template<int RM, int RN>
__global__ __launch_bounds__(256)
void gemm_wmma_kernel(const __bf16* __restrict__ A, const __bf16* __restrict__ W,
                      float* __restrict__ Cf, __bf16* __restrict__ Cb,
                      const float* __restrict__ resid,
                      int M, int N, int K) {
    const int lane = threadIdx.x & 31;
    const int wave = threadIdx.x >> 5;
    const int l16  = lane & 15;
    const int half = lane >> 4;

    const int tilesN = N / (16 * RN);
    const int nTiles = (M / (16 * RM)) * tilesN;
    const int wavesTotal = (gridDim.x * blockDim.x) >> 5;

    for (int t = blockIdx.x * 8 + wave; t < nTiles; t += wavesTotal) {
        const int m0 = (t / tilesN) * (16 * RM);
        const int n0 = (t % tilesN) * (16 * RN);

        v8f acc[RM][RN] = {};

        // A fragment: elems 0..7 -> K = half*8+0..7 ; elems 8..15 -> K = 16+half*8+0..7
        const __bf16* aptr[RM];
        #pragma unroll
        for (int r = 0; r < RM; ++r)
            aptr[r] = A + (size_t)(m0 + r * 16 + l16) * K + half * 8;
        // B fragment: elems 0..7 -> K = half*8+0..7 ; elems 8..15 -> K = 16+half*8+0..7
        const __bf16* bptr[RN];
        #pragma unroll
        for (int c = 0; c < RN; ++c)
            bptr[c] = W + (size_t)(n0 + c * 16 + l16) * K + half * 8;

        for (int k = 0; k < K; k += 32) {
            Frag fa[RM], fb[RN];
            #pragma unroll
            for (int r = 0; r < RM; ++r) {
                fa[r].q[0] = *reinterpret_cast<const u32x4*>(aptr[r] + k);
                fa[r].q[1] = *reinterpret_cast<const u32x4*>(aptr[r] + k + 16);
            }
            #pragma unroll
            for (int c = 0; c < RN; ++c) {
                fb[c].q[0] = *reinterpret_cast<const u32x4*>(bptr[c] + k);
                fb[c].q[1] = *reinterpret_cast<const u32x4*>(bptr[c] + k + 16);
            }
            #pragma unroll
            for (int r = 0; r < RM; ++r)
                #pragma unroll
                for (int c = 0; c < RN; ++c)
                    acc[r][c] = __builtin_amdgcn_wmma_f32_16x16x32_bf16(
                        false, fa[r].v, false, fb[c].v, (short)0, acc[r][c], false, false);
        }

        #pragma unroll
        for (int r = 0; r < RM; ++r) {
            #pragma unroll
            for (int c = 0; c < RN; ++c) {
                const int col = n0 + c * 16 + l16;
                #pragma unroll
                for (int i = 0; i < 8; ++i) {
                    const int rr = m0 + r * 16 + i + half * 8;  // D layout: vgpr i, lane half
                    const size_t o = (size_t)rr * N + col;
                    const float v = acc[r][c][i];
                    if (Cb)         Cb[o] = f2bf(v);
                    else if (resid) Cf[o] = v + resid[o];
                    else            Cf[o] = v;
                }
            }
        }
    }
}

// ---------------- causal depthwise conv1d (k=4) + SiLU ----------------
// xz: bf16 [MROWS, 2*D_INNER]; x_in = cols [0,D_INNER). Output u bf16 [MROWS, D_INNER].
__global__ void conv_silu_kernel(const __bf16* __restrict__ xz,
                                 const float* __restrict__ conv_w,
                                 const float* __restrict__ conv_b,
                                 __bf16* __restrict__ u) {
    const int total = MROWS * D_INNER;
    for (int i = blockIdx.x * blockDim.x + threadIdx.x; i < total; i += gridDim.x * blockDim.x) {
        const int d  = i & (D_INNER - 1);
        const int bl = i / D_INNER;
        const int l  = bl & (SEQLEN - 1);
        const int b  = bl / SEQLEN;
        float acc = conv_b[d];
        #pragma unroll
        for (int j = 0; j < D_CONV; ++j) {
            const int ls = l - (D_CONV - 1) + j;
            if (ls >= 0)
                acc += conv_w[d * D_CONV + j] *
                       bf2f(xz[((size_t)(b * SEQLEN + ls)) * (2 * D_INNER) + d]);
        }
        u[i] = f2bf(acc * sigmoidf_(acc));       // SiLU
    }
}

// ---------------- selective scan (sequential in L) ----------------
// One lane per (b, d) channel; h[16], A*log2e[16] in registers. B_t/C_t staged
// into LDS in LCHUNK-timestep chunks (2 barriers per chunk instead of per step).
__global__ __launch_bounds__(256)
void scan_kernel(const __bf16* __restrict__ dtraw, const float* __restrict__ dt_b,
                 const __bf16* __restrict__ u,     const float* __restrict__ xdbl,
                 const float* __restrict__ A_log,  const float* __restrict__ Dp,
                 const __bf16* __restrict__ xz,    __bf16* __restrict__ ybf) {
    const int b = blockIdx.x >> 3;                    // 8 blocks per batch
    const int d = ((blockIdx.x & 7) << 8) + threadIdx.x;
    __shared__ float bc[LCHUNK][2 * D_STATE];         // [li][0..15]=B, [16..31]=C

    float h[D_STATE], Ak[D_STATE];
    #pragma unroll
    for (int s = 0; s < D_STATE; ++s) {
        h[s]  = 0.f;
        Ak[s] = -__expf(A_log[d * D_STATE + s]) * 1.44269504088896340736f;  // A * log2(e)
    }
    const float dtbv = dt_b[d];
    const float Dv   = Dp[d];

    for (int l0 = 0; l0 < SEQLEN; l0 += LCHUNK) {
        __syncthreads();
        for (int i = threadIdx.x; i < LCHUNK * 2 * D_STATE; i += 256) {
            const int rl = i >> 5, c = i & 31;
            bc[rl][c] = xdbl[((size_t)(b * SEQLEN + l0 + rl)) * XPROJ_N + DT_RANK + c];
        }
        __syncthreads();

        for (int li = 0; li < LCHUNK; ++li) {
            const size_t row = (size_t)b * SEQLEN + l0 + li;
            const size_t idx = row * D_INNER + d;
            const float dtr = bf2f(dtraw[idx]) + dtbv;
            const float dt  = (dtr > 20.f) ? dtr : log1pf(__expf(dtr));   // softplus
            const float uu  = bf2f(u[idx]);
            const float du  = dt * uu;
            float y = 0.f;
            #pragma unroll
            for (int s = 0; s < D_STATE; ++s) {
                const float dA = exp2f(dt * Ak[s]);       // exp(dt*A)
                h[s] = dA * h[s] + du * bc[li][s];
                y += h[s] * bc[li][D_STATE + s];
            }
            float out = y + Dv * uu;
            const float z = bf2f(xz[row * (2 * D_INNER) + D_INNER + d]);
            out *= z * sigmoidf_(z);                      // gate with SiLU(z)
            ybf[idx] = f2bf(out);
        }
    }
}

// ---------------- host-side launch ----------------
static inline int gemm_blocks(int M, int N, int RM, int RN) {
    int tiles = (M / (16 * RM)) * (N / (16 * RN));
    return (tiles + 7) / 8;                            // 8 waves (one tile each) per block
}

extern "C" void kernel_launch(void* const* d_in, const int* in_sizes, int n_in,
                              void* d_out, int out_size, void* d_ws, size_t ws_size,
                              hipStream_t stream) {
    (void)in_sizes; (void)n_in; (void)out_size; (void)ws_size;
    const float* x        = (const float*)d_in[0];
    const float* norm_w   = (const float*)d_in[1];
    const float* in_proj  = (const float*)d_in[2];
    const float* conv_w   = (const float*)d_in[3];
    const float* conv_b   = (const float*)d_in[4];
    const float* x_proj   = (const float*)d_in[5];
    const float* dt_proj  = (const float*)d_in[6];
    const float* dt_pb    = (const float*)d_in[7];
    const float* A_log    = (const float*)d_in[8];
    const float* D_param  = (const float*)d_in[9];
    const float* out_proj = (const float*)d_in[10];
    float* out = (float*)d_out;

    // workspace sub-allocation (256B aligned)
    char* p = (char*)d_ws;
    auto alloc = [&](size_t bytes) { char* r = p; p += (bytes + 255) & ~(size_t)255; return r; };
    __bf16* xn_bf   = (__bf16*)alloc((size_t)MROWS * D_MODEL * 2);
    __bf16* w_in    = (__bf16*)alloc((size_t)2 * D_INNER * D_MODEL * 2);
    __bf16* w_xp    = (__bf16*)alloc((size_t)XPROJ_N * D_INNER * 2);
    __bf16* w_dt    = (__bf16*)alloc((size_t)D_INNER * DT_RANK * 2);
    __bf16* w_out   = (__bf16*)alloc((size_t)D_MODEL * D_INNER * 2);
    __bf16* xz_bf   = (__bf16*)alloc((size_t)MROWS * 2 * D_INNER * 2);
    __bf16* u_bf    = (__bf16*)alloc((size_t)MROWS * D_INNER * 2);
    float*  xdbl    = (float*) alloc((size_t)MROWS * XPROJ_N * 4);
    __bf16* dtA_bf  = (__bf16*)alloc((size_t)MROWS * DT_RANK * 2);
    __bf16* dtraw   = (__bf16*)alloc((size_t)MROWS * D_INNER * 2);
    __bf16* y_bf    = (__bf16*)alloc((size_t)MROWS * D_INNER * 2);

    // 1) weights -> bf16
    auto cvt = [&](const float* s, __bf16* d, int n) {
        int blocks = (n + 255) / 256; if (blocks > 2048) blocks = 2048;
        f32_to_bf16_kernel<<<blocks, 256, 0, stream>>>(s, d, n);
    };
    cvt(in_proj,  w_in,  2 * D_INNER * D_MODEL);
    cvt(x_proj,   w_xp,  XPROJ_N * D_INNER);
    cvt(dt_proj,  w_dt,  D_INNER * DT_RANK);
    cvt(out_proj, w_out, D_MODEL * D_INNER);

    // 2) RMSNorm -> bf16
    rmsnorm_kernel<<<MROWS, 256, 0, stream>>>(x, norm_w, xn_bf);

    // 3) in_proj GEMM: xz[8192,4096] (bf16), 64x64 wave tiles
    gemm_wmma_kernel<4, 4><<<gemm_blocks(MROWS, 2 * D_INNER, 4, 4), 256, 0, stream>>>(
        xn_bf, w_in, nullptr, xz_bf, nullptr, MROWS, 2 * D_INNER, D_MODEL);

    // 4) causal depthwise conv + SiLU -> u (bf16)
    conv_silu_kernel<<<4096, 256, 0, stream>>>(xz_bf, conv_w, conv_b, u_bf);

    // 5) x_proj GEMM: x_dbl[8192,96] (fp32; holds dt/B/C), 32x32 tiles (N=96)
    gemm_wmma_kernel<2, 2><<<gemm_blocks(MROWS, XPROJ_N, 2, 2), 256, 0, stream>>>(
        u_bf, w_xp, xdbl, nullptr, nullptr, MROWS, XPROJ_N, D_INNER);

    // 6) slice dt-rank cols -> bf16
    slice_dt_bf16_kernel<<<2048, 256, 0, stream>>>(xdbl, dtA_bf);

    // 7) dt_proj GEMM: dt_raw[8192,2048] (bf16; bias+softplus fused into scan)
    gemm_wmma_kernel<4, 4><<<gemm_blocks(MROWS, D_INNER, 4, 4), 256, 0, stream>>>(
        dtA_bf, w_dt, nullptr, dtraw, nullptr, MROWS, D_INNER, DT_RANK);

    // 8) selective scan (+ D*u skip, SiLU(z) gate) -> y (bf16)
    scan_kernel<<<BATCH * (D_INNER / 256), 256, 0, stream>>>(
        dtraw, dt_pb, u_bf, xdbl, A_log, D_param, xz_bf, y_bf);

    // 9) out_proj GEMM + residual -> d_out (fp32), 64x64 wave tiles
    gemm_wmma_kernel<4, 4><<<gemm_blocks(MROWS, D_MODEL, 4, 4), 256, 0, stream>>>(
        y_bf, w_out, out, nullptr, x, MROWS, D_MODEL, D_INNER);
}
